// WSWGAT_61125974557021
// MI455X (gfx1250) — matrix operand order repaired and checked by reading, not compiled, not run on hardware
//
#include <hip/hip_runtime.h>
#include <hip/hip_bf16.h>
#include <math.h>

// ---------------------------------------------------------------------------
// Problem constants (match reference)
// ---------------------------------------------------------------------------
#define NW_   100000
#define NS_   25000
#define E_    1600000
#define INDIM 128
#define OUTD  128
#define NH    8
#define HD    16
#define FFN_  512

typedef __bf16 bf16;
typedef __attribute__((ext_vector_type(16))) __bf16 bf16x16;
typedef __attribute__((ext_vector_type(8)))  float  floatx8;

// ---------------------------------------------------------------------------
// Packed-B layout ("fragment major"):
// For B[K,N] split into (K/32 x N/16) WMMA tiles, fragment f = kt*(N/16)+nt.
// Lane l (0..31) of a fragment holds 16 contiguous bf16:
//   b[j] = B[kt*32 + (l>>4)*16 + j][nt*16 + (l&15)],  j = 0..15
// so a B-fragment load is one 32-byte vector load per lane.
// ---------------------------------------------------------------------------
__device__ __forceinline__ int pack_src_index(int t, int N) {
    int j    = t & 15;
    int lane = (t >> 4) & 31;
    int f    = t >> 9;
    int NT   = N >> 4;
    int nt   = f % NT, kt = f / NT;
    int col  = nt * 16 + (lane & 15);
    int k    = kt * 32 + ((lane >> 4) << 4) + j;
    return k * N + col;
}

// ---------------------------------------------------------------------------
// Weight prep: pack Wfc (head-flattened 128x128), W1 (128x512), W2 (512x128)
// into fragment-major bf16; fold attention vectors: v[i][h] = sum_d Wfc[h,i,d]*a[h,d]
// ---------------------------------------------------------------------------
__global__ void k_convert(const float* __restrict__ Wfc,
                          const float* __restrict__ W1,
                          const float* __restrict__ W2,
                          const float* __restrict__ a_src,
                          const float* __restrict__ a_dst,
                          bf16* __restrict__ Wbpk, bf16* __restrict__ W1pk,
                          bf16* __restrict__ W2pk,
                          float* __restrict__ vsrc, float* __restrict__ vdst)
{
    const int nWb = INDIM * OUTD;        // 16384
    const int nW1 = OUTD * FFN_;         // 65536
    const int nW2 = FFN_ * OUTD;         // 65536
    const int total = nWb + nW1 + nW2 + 1024 + 1024;
    for (int idx = blockIdx.x * blockDim.x + threadIdx.x; idx < total;
         idx += gridDim.x * blockDim.x) {
        int t = idx;
        if (t < nWb) {
            int si  = pack_src_index(t, OUTD);
            int i   = si >> 7;            // row in 128x128  (input dim)
            int col = si & 127;           // h*16+d
            Wbpk[t] = (bf16)Wfc[(col >> 4) * (INDIM * HD) + i * HD + (col & 15)];
        } else if ((t -= nWb) < nW1) {
            W1pk[t] = (bf16)W1[pack_src_index(t, FFN_)];
        } else if ((t -= nW1) < nW2) {
            W2pk[t] = (bf16)W2[pack_src_index(t, OUTD)];
        } else if ((t -= nW2) < 1024) {
            int i = t >> 3, h = t & 7;
            float a = 0.f;
            for (int d = 0; d < HD; ++d)
                a += Wfc[h * (INDIM * HD) + i * HD + d] * a_src[h * HD + d];
            vsrc[i * NH + h] = a;
        } else {
            t -= 1024;
            int i = t >> 3, h = t & 7;
            float a = 0.f;
            for (int d = 0; d < HD; ++d)
                a += Wfc[h * (INDIM * HD) + i * HD + d] * a_dst[h * HD + d];
            vdst[i * NH + h] = a;
        }
    }
}

// ---------------------------------------------------------------------------
// WMMA GEMM: Out[M,N] = epilogue(A[M,K] fp32 -> bf16  x  Bpk (fragment-major bf16))
// One wave per block; wave computes a 32x64 tile: 2 A-frags x 4 B-frags = 8 WMMAs
// per 32-wide k-step. K is a compile-time constant so the k-loop unrolls and the
// scheduler pipelines loads across steps. Row guard is a clamp (no EXEC branch);
// out-of-range rows compute garbage that the guarded epilogue never stores.
// epilogue: +bias[col]; act==1 -> exact gelu; +res[row,col] if res != null.
// ---------------------------------------------------------------------------
template <int KK>
__global__ void k_wmma_gemm(const float* __restrict__ A,
                            const bf16*  __restrict__ Bpk,
                            const float* __restrict__ bias,
                            const float* __restrict__ res,
                            float* __restrict__ Out,
                            int M, int N, int act)
{
    const int lane = threadIdx.x & 31;
    const int hl   = lane >> 4;     // lane half
    const int ml   = lane & 15;     // row-in-tile (A) / col-in-tile (B,C)
    const int row0 = blockIdx.x * 32;
    const int n0   = blockIdx.y * 64;
    const int NT   = N >> 4;
    const int nt0  = n0 >> 4;

    floatx8 z8 = {0.f, 0.f, 0.f, 0.f, 0.f, 0.f, 0.f, 0.f};
    floatx8 acc[2][4];
#pragma unroll
    for (int rt = 0; rt < 2; ++rt)
#pragma unroll
        for (int t = 0; t < 4; ++t) acc[rt][t] = z8;

    const bf16x16* Bf = (const bf16x16*)Bpk;

    // clamped row indices (loop-invariant, no divergence)
    int rowc[2];
#pragma unroll
    for (int rt = 0; rt < 2; ++rt) {
        int r = row0 + rt * 16 + ml;
        rowc[rt] = r < M ? r : M - 1;
    }

#pragma unroll 4
    for (int k0 = 0; k0 < KK; k0 += 32) {
        const int kt = k0 >> 5;
        // ---- two A fragments (rows row0..row0+15, row0+16..row0+31)
        bf16x16 afr[2];
#pragma unroll
        for (int rt = 0; rt < 2; ++rt) {
            const float4* Ap = (const float4*)(A + (size_t)rowc[rt] * KK + k0);
            float4 q0 = Ap[hl * 4];         // k = hl*16   .. hl*16+3
            float4 q1 = Ap[hl * 4 + 1];     // k = hl*16+4 .. hl*16+7
            float4 q2 = Ap[hl * 4 + 2];     // k = hl*16+8 ..
            float4 q3 = Ap[hl * 4 + 3];
            afr[rt][0]  = (bf16)q0.x; afr[rt][1]  = (bf16)q0.y;
            afr[rt][2]  = (bf16)q0.z; afr[rt][3]  = (bf16)q0.w;
            afr[rt][4]  = (bf16)q1.x; afr[rt][5]  = (bf16)q1.y;
            afr[rt][6]  = (bf16)q1.z; afr[rt][7]  = (bf16)q1.w;
            afr[rt][8]  = (bf16)q2.x; afr[rt][9]  = (bf16)q2.y;
            afr[rt][10] = (bf16)q2.z; afr[rt][11] = (bf16)q2.w;
            afr[rt][12] = (bf16)q3.x; afr[rt][13] = (bf16)q3.y;
            afr[rt][14] = (bf16)q3.z; afr[rt][15] = (bf16)q3.w;
        }
        // ---- 4 B fragments, each shared by both A fragments
#pragma unroll
        for (int t = 0; t < 4; ++t) {
            bf16x16 b = Bf[((size_t)kt * NT + nt0 + t) * 32 + lane];
            acc[0][t] = __builtin_amdgcn_wmma_f32_16x16x32_bf16(
                false, afr[0], false, b, (short)0, acc[0][t], false, false);
            acc[1][t] = __builtin_amdgcn_wmma_f32_16x16x32_bf16(
                false, afr[1], false, b, (short)0, acc[1][t], false, false);
        }
    }

    // ---- epilogue + store. C/D layout: VGPR r -> M = r + 8*half, N = lane&15.
#pragma unroll
    for (int rt = 0; rt < 2; ++rt) {
#pragma unroll
        for (int t = 0; t < 4; ++t) {
            int col = n0 + t * 16 + ml;
#pragma unroll
            for (int r = 0; r < 8; ++r) {
                int orow = row0 + rt * 16 + r + hl * 8;
                if (orow < M) {
                    float v = acc[rt][t][r];
                    if (bias) v += bias[col];
                    if (act == 1) v = 0.5f * v * (1.0f + erff(v * 0.70710678118654752f));
                    if (res) v += res[(size_t)orow * N + col];
                    Out[(size_t)orow * N + col] = v;
                }
            }
        }
    }
}

// ---------------------------------------------------------------------------
// e[n,h] = dot(X[n,:], V[:,h])   (V stored [128][8])
// ---------------------------------------------------------------------------
__global__ void k_edot(const float* __restrict__ X, const float* __restrict__ V,
                       float* __restrict__ e, int Nrows)
{
    int idx = blockIdx.x * blockDim.x + threadIdx.x;
    if (idx >= Nrows * NH) return;
    int n = idx >> 3, h = idx & 7;
    const float* x = X + (size_t)n * INDIM;
    float acc = 0.f;
#pragma unroll 8
    for (int i = 0; i < INDIM; ++i) acc += x[i] * V[i * NH + h];
    e[idx] = acc;
}

// ---------------------------------------------------------------------------
// Zero-init: m_u (uint, monotone-encoded max), denom, attn accumulator
// ---------------------------------------------------------------------------
__global__ void k_init(unsigned* __restrict__ mu, float* __restrict__ denom,
                       float* __restrict__ attn)
{
    const int nm = NS_ * NH, nd = NS_ * NH, na = NS_ * OUTD;
    const int total = nm + nd + na;
    for (int idx = blockIdx.x * blockDim.x + threadIdx.x; idx < total;
         idx += gridDim.x * blockDim.x) {
        if (idx < nm) mu[idx] = 0u;                       // < encoding of any float
        else if (idx < nm + nd) denom[idx - nm] = 0.f;
        else attn[idx - nm - nd] = 0.f;
    }
}

__device__ __forceinline__ unsigned enc_f(float f) {
    unsigned b = __float_as_uint(f);
    return (b & 0x80000000u) ? ~b : (b | 0x80000000u);
}
__device__ __forceinline__ float dec_f(unsigned u) {
    unsigned b = (u & 0x80000000u) ? (u & 0x7FFFFFFFu) : ~u;
    return __uint_as_float(b);
}

// ---------------------------------------------------------------------------
// Edge pass A: score = leaky_relu(e_src[src]+e_dst[dst]); segment max (atomic)
// ---------------------------------------------------------------------------
__global__ void k_edgeA(const int* __restrict__ esrc, const int* __restrict__ edst,
                        const float* __restrict__ e_src, const float* __restrict__ e_dst,
                        float* __restrict__ score, unsigned* __restrict__ mu)
{
    int idx = blockIdx.x * blockDim.x + threadIdx.x;
    if (idx >= E_ * NH) return;
    int e = idx >> 3, h = idx & 7;
    int s = esrc[e], d = edst[e];
    float sc = e_src[s * NH + h] + e_dst[d * NH + h];
    sc = sc > 0.f ? sc : 0.01f * sc;
    score[idx] = sc;
    atomicMax(&mu[d * NH + h], enc_f(sc));
}

// ---------------------------------------------------------------------------
// Edge pass B: ex = exp(score - m[dst]); denom += ex (atomic)
// ---------------------------------------------------------------------------
__global__ void k_edgeB(const int* __restrict__ edst,
                        float* __restrict__ score,
                        const unsigned* __restrict__ mu,
                        float* __restrict__ denom)
{
    int idx = blockIdx.x * blockDim.x + threadIdx.x;
    if (idx >= E_ * NH) return;
    int e = idx >> 3, h = idx & 7;
    int d = edst[e];
    float m = dec_f(mu[d * NH + h]);
    if (!isfinite(m)) m = 0.f;          // nodes with no edges
    float ex = expf(score[idx] - m);
    score[idx] = ex;
    atomicAdd(&denom[d * NH + h], ex);
}

// ---------------------------------------------------------------------------
// Edge pass C: alpha = ex/denom; attn[dst,:] += alpha_h * z_w[src,:]
// 256 threads = 2 edges x 128 dims; head = dim/16. Coalesced gather, L2 atomics
// (attn target is 12.8 MB -> resident in 192 MB L2).
// ---------------------------------------------------------------------------
__global__ void k_edgeC(const int* __restrict__ esrc, const int* __restrict__ edst,
                        const float* __restrict__ ex, const float* __restrict__ denom,
                        const float* __restrict__ zw, float* __restrict__ attn)
{
    int e = blockIdx.x * 2 + (threadIdx.x >> 7);
    if (e >= E_) return;
    int j = threadIdx.x & 127;
    int h = j >> 4;
    int s = esrc[e], d = edst[e];
    float alpha = ex[e * NH + h] / fmaxf(denom[d * NH + h], 1e-9f);
    atomicAdd(&attn[(size_t)d * OUTD + j], alpha * zw[(size_t)s * OUTD + j]);
}

// ---------------------------------------------------------------------------
// h = elu(attn) + s;  xn = LayerNorm(h)*gamma+beta.  One 128-thread block/row.
// ---------------------------------------------------------------------------
__global__ void k_fuse(const float* __restrict__ attn, const float* __restrict__ s,
                       const float* __restrict__ gamma, const float* __restrict__ beta,
                       float* __restrict__ hbuf, float* __restrict__ xnbuf)
{
    int n = blockIdx.x, j = threadIdx.x;
    float o  = attn[(size_t)n * OUTD + j];
    float hv = (o > 0.f ? o : expm1f(o)) + s[(size_t)n * OUTD + j];

    __shared__ float r1[128], r2[128];
    r1[j] = hv; r2[j] = hv * hv;
    __syncthreads();
    for (int st = 64; st > 0; st >>= 1) {
        if (j < st) { r1[j] += r1[j + st]; r2[j] += r2[j + st]; }
        __syncthreads();
    }
    float mu  = r1[0] * (1.f / 128.f);
    float var = r2[0] * (1.f / 128.f) - mu * mu;
    float xn  = (hv - mu) * rsqrtf(var + 1e-6f) * gamma[j] + beta[j];

    hbuf [(size_t)n * OUTD + j] = hv;
    xnbuf[(size_t)n * OUTD + j] = xn;
}

// ---------------------------------------------------------------------------
// Host-side launch
// ---------------------------------------------------------------------------
extern "C" void kernel_launch(void* const* d_in, const int* in_sizes, int n_in,
                              void* d_out, int out_size, void* d_ws, size_t ws_size,
                              hipStream_t stream)
{
    const float* w     = (const float*)d_in[0];
    const float* s     = (const float*)d_in[1];
    const int*   esrc  = (const int*)  d_in[2];
    const int*   edst  = (const int*)  d_in[3];
    const float* Wfc   = (const float*)d_in[4];
    const float* a_src = (const float*)d_in[5];
    const float* a_dst = (const float*)d_in[6];
    const float* W1    = (const float*)d_in[7];
    const float* b1    = (const float*)d_in[8];
    const float* W2    = (const float*)d_in[9];
    const float* b2    = (const float*)d_in[10];
    const float* gamma = (const float*)d_in[11];
    const float* beta  = (const float*)d_in[12];
    float* out = (float*)d_out;

    // Workspace carve-up (256B aligned)
    char* ws = (char*)d_ws;
    size_t off = 0;
    auto carve = [&](size_t bytes) -> char* {
        char* p = ws + off;
        off += (bytes + 255) & ~(size_t)255;
        return p;
    };
    bf16*     Wbpk  = (bf16*)    carve((size_t)INDIM * OUTD * 2);
    bf16*     W1pk  = (bf16*)    carve((size_t)OUTD * FFN_ * 2);
    bf16*     W2pk  = (bf16*)    carve((size_t)FFN_ * OUTD * 2);
    float*    vsrc  = (float*)   carve((size_t)INDIM * NH * 4);
    float*    vdst  = (float*)   carve((size_t)INDIM * NH * 4);
    float*    zw    = (float*)   carve((size_t)NW_ * OUTD * 4);
    float*    e_src = (float*)   carve((size_t)NW_ * NH * 4);
    float*    e_dst = (float*)   carve((size_t)NS_ * NH * 4);
    float*    score = (float*)   carve((size_t)E_ * NH * 4);
    unsigned* m_u   = (unsigned*)carve((size_t)NS_ * NH * 4);
    float*    denom = (float*)   carve((size_t)NS_ * NH * 4);
    float*    attn  = (float*)   carve((size_t)NS_ * OUTD * 4);
    float*    hbuf  = (float*)   carve((size_t)NS_ * OUTD * 4);
    float*    xnbuf = (float*)   carve((size_t)NS_ * OUTD * 4);
    float*    inter = (float*)   carve((size_t)NS_ * FFN_ * 4);
    (void)ws_size; (void)in_sizes; (void)n_in; (void)out_size;

    // 1) weight prep (pack to fragment-major bf16)
    k_convert<<<256, 256, 0, stream>>>(Wfc, W1, W2, a_src, a_dst,
                                       Wbpk, W1pk, W2pk, vsrc, vdst);
    // 2) z_w = w @ Wfc (WMMA bf16), 32x64 tile per wave
    k_wmma_gemm<INDIM><<<dim3(NW_ / 32, OUTD / 64), 32, 0, stream>>>(
        w, Wbpk, nullptr, nullptr, zw, NW_, OUTD, 0);
    // 3) attention logits per node
    k_edot<<<(NW_ * NH + 255) / 256, 256, 0, stream>>>(w, vsrc, e_src, NW_);
    k_edot<<<(NS_ * NH + 255) / 256, 256, 0, stream>>>(s, vdst, e_dst, NS_);
    // 4) init segment buffers
    k_init<<<2048, 256, 0, stream>>>(m_u, denom, attn);
    // 5) segment softmax over edges
    k_edgeA<<<(E_ * NH) / 256, 256, 0, stream>>>(esrc, edst, e_src, e_dst, score, m_u);
    k_edgeB<<<(E_ * NH) / 256, 256, 0, stream>>>(edst, score, m_u, denom);
    // 6) weighted message scatter
    k_edgeC<<<(E_ + 1) / 2, 256, 0, stream>>>(esrc, edst, score, denom, zw, attn);
    // 7) elu + residual + LayerNorm
    k_fuse<<<NS_, 128, 0, stream>>>(attn, s, gamma, beta, hbuf, xnbuf);
    // 8) FFN: inter = gelu(xn @ W1 + b1)
    k_wmma_gemm<OUTD><<<dim3((NS_ + 31) / 32, FFN_ / 64), 32, 0, stream>>>(
        xnbuf, W1pk, b1, nullptr, inter, NS_, FFN_, 1);
    // 9) out = inter @ W2 + b2 + h
    k_wmma_gemm<FFN_><<<dim3((NS_ + 31) / 32, OUTD / 64), 32, 0, stream>>>(
        inter, W2pk, b2, hbuf, out, NS_, OUTD, 0);
}
